// FrequencyGuidedAttention_9904194585068
// MI455X (gfx1250) — matrix operand order repaired and checked by reading, not compiled, not run on hardware
//
#include <hip/hip_runtime.h>
#include <hip/hip_bf16.h>

// ---------------------------------------------------------------------------
// FrequencyGuidedAttention for MI455X (gfx1250, wave32, WMMA + async/TDM)
// B=4, C=96, H=W=256, NH=4  (head channels = 24, HW = 65536)
// ---------------------------------------------------------------------------

static constexpr int kB  = 4;
static constexpr int kC  = 96;
static constexpr int kNH = 4;
static constexpr int kCH = 24;          // channels per head
static constexpr int kH  = 256;
static constexpr int kW  = 256;
static constexpr int kHW = kH * kW;     // 65536

typedef __attribute__((ext_vector_type(16))) _Float16 v16h;
typedef __attribute__((ext_vector_type(8)))  float    v8f;
typedef __attribute__((ext_vector_type(4)))  unsigned int v4u;
typedef __attribute__((ext_vector_type(8)))  int          v8i_;
typedef __attribute__((ext_vector_type(4)))  int          v4i_;

// Flat addresses in the LDS aperture map to LDS via addr[31:0] (ISA 10.2).
__device__ __forceinline__ unsigned lds_addr_of(const void* p) {
    return (unsigned)(size_t)p;
}

// Per-lane async DMA: 16 bytes global -> LDS, tracked by ASYNCcnt.
__device__ __forceinline__ void async_copy_b128(unsigned lds_byte, const void* g) {
    asm volatile("global_load_async_to_lds_b128 %0, %1, off"
                 :: "v"(lds_byte), "v"(g) : "memory");
}
__device__ __forceinline__ void wait_async0() {
    asm volatile("s_wait_asynccnt 0x0" ::: "memory");
}

// One-shot TDM DMA of a contiguous run of f16 elements into LDS (1-D tile).
// amdgpu-toolchain (clang-23) 6-arg form:
//   (uint32x4 g0, int32x8 g1, int32x4 g2, int32x4 g3, int32x8 g4, i32 cpol)
__device__ __forceinline__ void tdm_load_1d_f16(unsigned lds_byte, const void* gsrc,
                                                unsigned nelem /* < 65536 */) {
#if __has_builtin(__builtin_amdgcn_tensor_load_to_lds)
    const unsigned long long ga = (unsigned long long)(size_t)gsrc;
    // D# group 0: count=1 (valid user descriptor), lds_addr, global_addr, type=2
    v4u g0 = { 1u,
               lds_byte,
               (unsigned)(ga & 0xffffffffu),
               (unsigned)((ga >> 32) & 0x01ffffffu) | (2u << 30) };
    // D# group 1: data_size=1 (2B); tensor_dim0=nelem; tensor_dim1=1;
    //             tile_dim0=nelem; tile_dim1=1; dim0 stride=nelem
    v8i_ g1 = { (int)(1u << 16),
                (int)((nelem & 0xffffu) << 16),
                (int)(((nelem >> 16) & 0xffffu) | (1u << 16)),
                (int)((nelem & 0xffffu) << 16),
                1,
                (int)nelem,
                0, 0 };
    v4i_ z4 = { 0, 0, 0, 0 };
    v8i_ z8 = { 0, 0, 0, 0, 0, 0, 0, 0 };
    __builtin_amdgcn_tensor_load_to_lds(g0, g1, z4, z4, z8, 0);
    __builtin_amdgcn_s_wait_tensorcnt(0);
#else
    (void)lds_byte; (void)gsrc; (void)nelem;
#endif
}

// ===========================================================================
// Channel GEMM: Y[b][oc][p] = sum_ic W[oc][ic] * X[b][ic][p]
// W (COUT x 96) in LDS f16, X tile (96 x 128 positions) in LDS f16,
// v_wmma_f32_16x16x32_f16.  256 threads = 8 waves; wave w owns N-tile w*16.
// f16 inputs are staged with async-to-LDS / TDM; f32 inputs go through VGPRs
// (they need conversion anyway).
// ===========================================================================
template<int COUT, typename TIN, typename TW, typename TOUT, bool W_PER_BATCH>
__global__ __launch_bounds__(256)
void chan_gemm_kernel(const TIN* __restrict__ X, const TW* __restrict__ W,
                      TOUT* __restrict__ Y)
{
    __shared__ _Float16 Ws[COUT * 96];     // <= 36 KB
    __shared__ _Float16 Xs[96 * 128];      // 24 KB  (layout: c*128 + p)

    const int tid   = threadIdx.x;
    const int b     = blockIdx.y;
    const int pbase = blockIdx.x * 128;

    const TW* Wsrc = W_PER_BATCH ? (W + (size_t)b * COUT * 96) : W;
    const TIN* Xb  = X + (size_t)b * 96 * kHW + pbase;

    if constexpr (W_PER_BATCH && sizeof(TW) == 2) {
#if __has_builtin(__builtin_amdgcn_tensor_load_to_lds)
        if (tid < 32)                       // wave 0 issues the TDM DMA
            tdm_load_1d_f16(lds_addr_of(&Ws[0]), Wsrc, COUT * 96);
#else
        for (int i = tid; i < COUT * 96; i += 256)
            Ws[i] = (_Float16)(float)Wsrc[i];
#endif
    } else {
        for (int i = tid; i < COUT * 96; i += 256)
            Ws[i] = (_Float16)(float)Wsrc[i];
    }

    if constexpr (sizeof(TIN) == 2) {
        // f16 input: per-lane async DMA, 16B chunks (1536 chunks total)
        const unsigned lb = lds_addr_of(&Xs[0]);
#pragma unroll
        for (int j = 0; j < 6; ++j) {
            const int idx = tid + j * 256;
            const int row = idx >> 4, ch = idx & 15;   // 16 x b128 per row
            async_copy_b128(lb + (unsigned)(row * 128 + ch * 8) * 2,
                            (const void*)(Xb + (size_t)row * kHW + ch * 8));
        }
        wait_async0();
    } else {
        for (int i = tid; i < 96 * 128; i += 256) {
            const int c = i >> 7, p = i & 127;
            Xs[i] = (_Float16)(float)Xb[(size_t)c * kHW + p];
        }
    }
    __syncthreads();

    const int wv  = tid >> 5;
    const int ln  = tid & 31;
    const int s   = ln >> 4;        // half-wave select
    const int l15 = ln & 15;
    const int nb  = wv * 16;        // N-tile base (position within 128-tile)

    for (int mt = 0; mt < COUT / 16; ++mt) {
        v8f acc = {};
#pragma unroll
        for (int kt = 0; kt < 3; ++kt) {     // K = 96 = 3 x 32
            const int kb = kt * 32;
            v16h a, bb;
#pragma unroll
            for (int r = 0; r < 8; ++r) {
                // A (16x32 f16) layout: lane m=l15; VGPR r holds K pair
                const int kk = (r < 4 ? 2 * r : 16 + 2 * (r - 4)) + 8 * s;
                a[2 * r]     = Ws[(mt * 16 + l15) * 96 + kb + kk];
                a[2 * r + 1] = Ws[(mt * 16 + l15) * 96 + kb + kk + 1];
                // B (32x16 f16) layout: lane n=l15, K = same as A (kk + j)
                const int kp = kb + kk;
                bb[2 * r]     = Xs[kp * 128 + nb + l15];
                bb[2 * r + 1] = Xs[(kp + 1) * 128 + nb + l15];
            }
            acc = __builtin_amdgcn_wmma_f32_16x16x32_f16(
                false, a, false, bb, (short)0, acc, false, false);
        }
        // C/D (16x16 f32): VGPR r -> row r + 8*s, col l15
#pragma unroll
        for (int r = 0; r < 8; ++r) {
            const int m = mt * 16 + r + 8 * s;
            Y[((size_t)(b * COUT + m)) * kHW + pbase + nb + l15] = (TOUT)acc[r];
        }
    }
}

// ===========================================================================
// Fused: depthwise 3x3 (zero 'SAME' pad) for q,k,v of ONE head on a 16x16
// spatial tile + per-channel sum(q^2)/sum(k^2) + unnormalized Gram q.k^T
// (24x24 per head, WMMA over K=256 positions, f32 atomic accumulate),
// and v written to HBM in f16.  q/k never touch HBM.
// Grid: (256 tiles, NH, B); 256 threads.
// ===========================================================================
__global__ __launch_bounds__(256)
void dw_gram_kernel(const _Float16* __restrict__ yq,
                    const _Float16* __restrict__ ykv,
                    const float* __restrict__ wqd,   // [96][9]
                    const float* __restrict__ wkvd,  // [192][9]
                    float* __restrict__ gram,        // [B][NH][24][24]
                    float* __restrict__ sqq,         // [B][96]
                    float* __restrict__ sqk,         // [B][96]
                    _Float16* __restrict__ vout)     // [B][96][HW]
{
    __shared__ _Float16 halo[24 * 18 * 18];   // staged input tile (+1 halo)
    __shared__ _Float16 qs[32 * 256];         // rows 24..31 zero padded
    __shared__ _Float16 ks[32 * 256];
    __shared__ float    dww[24 * 9];
    __shared__ float    sqred[24];

    const int tid = threadIdx.x;
    const int t   = blockIdx.x;
    const int h   = blockIdx.y;
    const int b   = blockIdx.z;
    const int tx0 = (t & 15) * 16;
    const int ty0 = (t >> 4) * 16;
    const int c0  = h * kCH;

    auto load_halo = [&](const _Float16* src, int cbase, int Csrc) {
        for (int i = tid; i < 24 * 324; i += 256) {
            const int c = i / 324, rem = i % 324, yy = rem / 18, xx = rem % 18;
            const int gy = ty0 - 1 + yy, gx = tx0 - 1 + xx;
            _Float16 v = (_Float16)0.f;
            if (gy >= 0 && gy < kH && gx >= 0 && gx < kW)
                v = src[((size_t)(b * Csrc + cbase + c)) * kHW + gy * kW + gx];
            halo[i] = v;
        }
    };
    auto dwconv_to = [&](_Float16* dst, bool do_sq) {
        for (int i = tid; i < 24 * 256; i += 256) {
            const int c = i >> 8, p = i & 255;
            const int py = p >> 4, px = p & 15;
            float acc = 0.f;
#pragma unroll
            for (int dy = 0; dy < 3; ++dy)
#pragma unroll
                for (int dx = 0; dx < 3; ++dx)
                    acc += (float)halo[c * 324 + (py + dy) * 18 + (px + dx)] *
                           dww[c * 9 + dy * 3 + dx];
            dst[c * 256 + p] = (_Float16)acc;
            if (do_sq) atomicAdd(&sqred[c], acc * acc);
        }
        for (int i = tid + 24 * 256; i < 32 * 256; i += 256)
            dst[i] = (_Float16)0.f;   // zero pad rows 24..31
    };

    // ---------------- Q ----------------
    for (int i = tid; i < 24 * 9; i += 256) dww[i] = wqd[c0 * 9 + i];
    load_halo(yq, c0, 96);
    if (tid < 24) sqred[tid] = 0.f;
    __syncthreads();
    dwconv_to(qs, true);
    __syncthreads();
    if (tid < 24) { atomicAdd(&sqq[b * 96 + c0 + tid], sqred[tid]); sqred[tid] = 0.f; }
    // ---------------- K ----------------
    for (int i = tid; i < 24 * 9; i += 256) dww[i] = wkvd[c0 * 9 + i];
    load_halo(ykv, c0, 192);
    __syncthreads();
    dwconv_to(ks, true);
    __syncthreads();
    if (tid < 24) atomicAdd(&sqk[b * 96 + c0 + tid], sqred[tid]);
    // ---------------- V staging (halo reuse) ----------------
    for (int i = tid; i < 24 * 9; i += 256) dww[i] = wkvd[(96 + c0) * 9 + i];
    load_halo(ykv, 96 + c0, 192);
    __syncthreads();

    // ---------------- Gram = Q(24x256) * K(24x256)^T via WMMA --------------
    const int wv = tid >> 5, ln = tid & 31, s = ln >> 4, l15 = ln & 15;
    if (wv < 4) {                         // wave-uniform: EXEC all ones
        const int mt = wv >> 1, nt = wv & 1;
        v8f acc = {};
        for (int kt = 0; kt < 8; ++kt) { // 256 positions = 8 x K32
            const int kb = kt * 32;
            v16h a, bb;
#pragma unroll
            for (int r = 0; r < 8; ++r) {
                const int kk = (r < 4 ? 2 * r : 16 + 2 * (r - 4)) + 8 * s;
                a[2 * r]     = qs[(mt * 16 + l15) * 256 + kb + kk];
                a[2 * r + 1] = qs[(mt * 16 + l15) * 256 + kb + kk + 1];
                const int kp = kb + kk;                // B[k][n] = K[n][k]
                bb[2 * r]     = ks[(nt * 16 + l15) * 256 + kp];
                bb[2 * r + 1] = ks[(nt * 16 + l15) * 256 + kp + 1];
            }
            acc = __builtin_amdgcn_wmma_f32_16x16x32_f16(
                false, a, false, bb, (short)0, acc, false, false);
        }
#pragma unroll
        for (int r = 0; r < 8; ++r) {
            const int m = mt * 16 + r + 8 * s;
            const int n = nt * 16 + l15;
            if (m < kCH && n < kCH)
                atomicAdd(&gram[((size_t)(b * kNH + h) * kCH + m) * kCH + n], acc[r]);
        }
    }

    // ---------------- V dwconv -> HBM (f16) --------------------------------
    for (int i = tid; i < 24 * 256; i += 256) {
        const int c = i >> 8, p = i & 255;
        const int py = p >> 4, px = p & 15;
        float acc = 0.f;
#pragma unroll
        for (int dy = 0; dy < 3; ++dy)
#pragma unroll
            for (int dx = 0; dx < 3; ++dx)
                acc += (float)halo[c * 324 + (py + dy) * 18 + (px + dx)] *
                       dww[c * 9 + dy * 3 + dx];
        vout[((size_t)(b * 96 + c0 + c)) * kHW + (ty0 + py) * kW + tx0 + px] =
            (_Float16)acc;
    }
}

// ===========================================================================
// Softmax over normalized Gram + fold attention into output projection:
//   attn[h] = softmax( G[h] / (|q_c||k_d|) * T[h] )
//   M[b]    = W_out * blockdiag(attn[h])        (96x96 f16)
// One block per batch.
// ===========================================================================
__global__ __launch_bounds__(256)
void softmax_m_kernel(const float* __restrict__ gram,
                      const float* __restrict__ sqq,
                      const float* __restrict__ sqk,
                      const float* __restrict__ temp,   // [NH]
                      const float* __restrict__ w_out,  // [96][96]
                      _Float16* __restrict__ Mmat)      // [B][96][96]
{
    __shared__ float attn[kNH * kCH * kCH];
    const int b = blockIdx.x, tid = threadIdx.x;

    for (int i = tid; i < kNH * kCH * kCH; i += 256) {
        const int h = i / (kCH * kCH), r = i % (kCH * kCH);
        const int c = r / kCH, d = r % kCH;
        const float nq = fmaxf(sqrtf(sqq[b * 96 + h * kCH + c]), 1e-12f);
        const float nk = fmaxf(sqrtf(sqk[b * 96 + h * kCH + d]), 1e-12f);
        attn[i] = gram[(size_t)(b * kNH + h) * kCH * kCH + r] / (nq * nk) * temp[h];
    }
    __syncthreads();
    if (tid < kNH * kCH) {               // one thread per (head, row)
        float* row = &attn[(tid / kCH) * kCH * kCH + (tid % kCH) * kCH];
        float mx = -3.0e38f;
        for (int d = 0; d < kCH; ++d) mx = fmaxf(mx, row[d]);
        float sum = 0.f;
        for (int d = 0; d < kCH; ++d) { row[d] = __expf(row[d] - mx); sum += row[d]; }
        const float inv = 1.f / sum;
        for (int d = 0; d < kCH; ++d) row[d] *= inv;
    }
    __syncthreads();
    for (int i = tid; i < 96 * 96; i += 256) {
        const int oc = i / 96, ic = i % 96;
        const int h = ic / kCH, d = ic % kCH;
        float acc = 0.f;
        for (int c = 0; c < kCH; ++c)
            acc += w_out[oc * 96 + h * kCH + c] * attn[h * kCH * kCH + c * kCH + d];
        Mmat[((size_t)b * 96 + oc) * 96 + ic] = (_Float16)acc;
    }
}

// ===========================================================================
extern "C" void kernel_launch(void* const* d_in, const int* in_sizes, int n_in,
                              void* d_out, int out_size, void* d_ws, size_t ws_size,
                              hipStream_t stream)
{
    (void)in_sizes; (void)n_in; (void)out_size; (void)ws_size;

    const float* x     = (const float*)d_in[0];
    const float* ff    = (const float*)d_in[1];
    const float* w_q   = (const float*)d_in[2];
    const float* w_kv  = (const float*)d_in[3];
    const float* w_qd  = (const float*)d_in[4];
    const float* w_kvd = (const float*)d_in[5];
    const float* w_out = (const float*)d_in[6];
    const float* temp  = (const float*)d_in[7];
    float* out = (float*)d_out;

    char* ws = (char*)d_ws;
    size_t o = 0;
    _Float16* yq  = (_Float16*)(ws + o); o += (size_t)kB *  96 * kHW * 2;
    _Float16* ykv = (_Float16*)(ws + o); o += (size_t)kB * 192 * kHW * 2;
    _Float16* vbf = (_Float16*)(ws + o); o += (size_t)kB *  96 * kHW * 2;
    const size_t zoff = o;
    float* gram = (float*)(ws + o); o += (size_t)kB * kNH * kCH * kCH * 4;
    float* sqq  = (float*)(ws + o); o += (size_t)kB * 96 * 4;
    float* sqk  = (float*)(ws + o); o += (size_t)kB * 96 * 4;
    const size_t zbytes = o - zoff;
    _Float16* Mm = (_Float16*)(ws + o); o += (size_t)kB * 96 * 96 * 2;

    (void)hipMemsetAsync(ws + zoff, 0, zbytes, stream);

    dim3 gGemm(kHW / 128, kB);
    // q path 1x1 conv:  yq = w_q @ ff
    chan_gemm_kernel<96,  float, float, _Float16, false>
        <<<gGemm, 256, 0, stream>>>(ff, w_q, yq);
    // kv path 1x1 conv: ykv = w_kv @ x
    chan_gemm_kernel<192, float, float, _Float16, false>
        <<<gGemm, 256, 0, stream>>>(x, w_kv, ykv);
    // fused dwconv + sumsq + Gram + v
    dw_gram_kernel<<<dim3(256, kNH, kB), 256, 0, stream>>>(
        yq, ykv, w_qd, w_kvd, gram, sqq, sqk, vbf);
    // softmax + fold into output projection
    softmax_m_kernel<<<dim3(kB), 256, 0, stream>>>(gram, sqq, sqk, temp, w_out, Mm);
    // out = M[b] @ V[b] (f16 data: async-to-LDS + TDM staging)
    chan_gemm_kernel<96, _Float16, _Float16, float, true>
        <<<gGemm, 256, 0, stream>>>(vbf, Mm, out);
}
